// QAgent_7129645711381
// MI455X (gfx1250) — compile-verified, matrix operand order checked by reading, not run on hardware
//
#include <hip/hip_runtime.h>
#include <math.h>

// ---------------- problem constants (from the reference) ----------------
#define TT      4096          // timesteps
#define BKB     32            // batch
#define OBSD    3
#define DM      256           // d_model
#define DIN     512           // d_inner
#define DST     64            // d_state
#define NH      8             // heads
#define HD      64            // headdim
#define NCH     64            // chunks (T / 64)
#define CONVD   640           // d_inner + 2*d_state
#define DIP     1160          // in_proj rows
#define TOK     (TT * BKB)    // 131072 tokens

typedef __attribute__((ext_vector_type(16))) _Float16 v16h;
typedef __attribute__((ext_vector_type(8)))  float    v8f;

// ---------------- WMMA helpers (CDNA5 16x16x32 f16 layouts) ----------------
// K index held by half j of a lane's 16-half fragment (hi = lane>=16).
// Pattern is two contiguous 16B runs per lane -> compiler emits ds_load_b128 pairs.
__device__ __forceinline__ int kmap(int j, int hi) {
  int v = j >> 1, p = j & 1;
  int b = (v < 4) ? (2 * v) : (16 + 2 * (v - 4));
  return b + p + 8 * hi;
}

// operand stored row-major [frag_row][K] in LDS
__device__ __forceinline__ v16h frag_rowmajor(const _Float16* lds, int row,
                                              int stride, int koff, int hi) {
  v16h f;
#pragma unroll
  for (int j = 0; j < 16; ++j) f[j] = lds[row * stride + koff + kmap(j, hi)];
  return f;
}

// operand stored K-major [K][frag_col] in LDS (free transpose)
__device__ __forceinline__ v16h frag_kmajor(const _Float16* lds, int col,
                                            int stride, int koff, int hi) {
  v16h f;
#pragma unroll
  for (int j = 0; j < 16; ++j) f[j] = lds[(koff + kmap(j, hi)) * stride + col];
  return f;
}

__device__ __forceinline__ v8f wmma16(v16h a, v16h b, v8f c) {
  return __builtin_amdgcn_wmma_f32_16x16x32_f16(false, a, false, b,
                                                (short)0, c, false, false);
}

// ---------------- CDNA5 async global->LDS (ASYNCcnt path) ----------------
// 16B per lane, written directly into LDS with no VGPR round trip.
__device__ __forceinline__ void async_b128(const void* gaddr, unsigned lds_off) {
  asm volatile("global_load_async_to_lds_b128 %0, %1, off"
               :: "v"(lds_off), "v"((unsigned long long)(size_t)gaddr)
               : "memory");
}
__device__ __forceinline__ void wait_async0() {
  asm volatile("s_wait_asynccnt 0x0" ::: "memory");
}
__device__ __forceinline__ void wait_async3() {
  asm volatile("s_wait_asynccnt 0x3" ::: "memory");
}

__device__ __forceinline__ float softplusf(float x) {
  return (x > 20.f) ? x : log1pf(expf(x));
}
__device__ __forceinline__ float siluf(float x) {
  return x / (1.f + expf(-x));
}

// ---------------- 1) convert weights to f16 ----------------
__global__ void cvt_weights(const float* __restrict__ wi, _Float16* __restrict__ wih, int ni,
                            const float* __restrict__ wo, _Float16* __restrict__ woh, int no) {
  int idx = blockIdx.x * 256 + threadIdx.x;
  if (idx < ni) wih[idx] = (_Float16)wi[idx];
  if (idx < no) woh[idx] = (_Float16)wo[idx];
}

// ---------------- 2) embed: u[bk*T+t][d] = obs @ W_in^T + b_in ----------------
__global__ void embed_k(const float* __restrict__ obs, const float* __restrict__ Wi,
                        const float* __restrict__ bi, _Float16* __restrict__ u) {
  size_t idx = (size_t)blockIdx.x * 256 + threadIdx.x;    // token*256 + d
  int d = (int)(idx & 255);
  size_t tok = idx >> 8;
  int t  = (int)(tok % TT);
  int bk = (int)(tok / TT);
  const float* o = obs + ((size_t)t * BKB + bk) * OBSD;
  float v = bi[d] + o[0] * Wi[d * 3 + 0] + o[1] * Wi[d * 3 + 1] + o[2] * Wi[d * 3 + 2];
  u[idx] = (_Float16)v;
}

// ---------------- 3) WMMA GEMM: C[M,N] = A[M,K] * B[N,K]^T ----------------
// 128x64 block tile, 8 waves; each wave owns a 32x32 macro-tile (4 WMMA accumulators).
// Double-buffered LDS filled by async global->LDS copies.
__device__ __forceinline__ void stage_tiles(const _Float16* A, const _Float16* B,
                                            int N, int K, int bm, int bn, int k0,
                                            _Float16* sAbuf, _Float16* sBbuf, int tid) {
  // A tile: 128 rows x 32 halves = 512 x b128 -> 2 pieces per thread
  int p0 = tid, p1 = tid + 256;
  int r0 = p0 >> 2, s0 = p0 & 3;
  int r1 = p1 >> 2, s1 = p1 & 3;
  async_b128(A + (size_t)(bm + r0) * K + k0 + s0 * 8,
             (unsigned)(size_t)(sAbuf + r0 * 32 + s0 * 8));
  async_b128(A + (size_t)(bm + r1) * K + k0 + s1 * 8,
             (unsigned)(size_t)(sAbuf + r1 * 32 + s1 * 8));
  // B tile: 64 rows x 32 halves = 256 x b128 -> 1 piece per thread (row clamped;
  // out-of-range output columns are masked at the store)
  int r = tid >> 2, s = tid & 3;
  int gn = bn + r; if (gn > N - 1) gn = N - 1;
  async_b128(B + (size_t)gn * K + k0 + s * 8,
             (unsigned)(size_t)(sBbuf + r * 32 + s * 8));
}

__global__ void __launch_bounds__(256)
gemm_wmma(const _Float16* __restrict__ A, const _Float16* __restrict__ B,
          float* __restrict__ Cf, _Float16* __restrict__ Ch,
          int M, int N, int K, int ldc, int out_half) {
  __shared__ __align__(16) _Float16 sA[2][128 * 32];
  __shared__ __align__(16) _Float16 sB[2][64 * 32];
  const int tid = threadIdx.x;
  const int bm = blockIdx.y * 128;
  const int bn = blockIdx.x * 64;
  const int w = tid >> 5, lane = tid & 31, hi = lane >> 4, lo = lane & 15;
  const int wr = w & 3;          // 4 row-groups of 32 rows
  const int wc = w >> 2;         // 2 col-groups of 32 cols
  v8f acc00 = {}, acc01 = {}, acc10 = {}, acc11 = {};

  stage_tiles(A, B, N, K, bm, bn, 0, sA[0], sB[0], tid);
  int buf = 0;
  for (int k0 = 0; k0 < K; k0 += 32, buf ^= 1) {
    if (k0 + 32 < K) {
      stage_tiles(A, B, N, K, bm, bn, k0 + 32, sA[buf ^ 1], sB[buf ^ 1], tid);
      wait_async3();             // current buffer's 3 copies have landed
    } else {
      wait_async0();
    }
    __syncthreads();
    v16h a0 = frag_rowmajor(sA[buf], wr * 32 + lo, 32, 0, hi);
    v16h a1 = frag_rowmajor(sA[buf], wr * 32 + 16 + lo, 32, 0, hi);
    v16h b0 = frag_rowmajor(sB[buf], wc * 32 + lo, 32, 0, hi);
    v16h b1 = frag_rowmajor(sB[buf], wc * 32 + 16 + lo, 32, 0, hi);
    acc00 = wmma16(a0, b0, acc00);
    acc01 = wmma16(a0, b1, acc01);
    acc10 = wmma16(a1, b0, acc10);
    acc11 = wmma16(a1, b1, acc11);
    __syncthreads();
  }
#pragma unroll
  for (int r = 0; r < 8; ++r) {
    int m0 = bm + wr * 32 + r + 8 * hi;
    int m1 = m0 + 16;
    int n0 = bn + wc * 32 + lo;
    int n1 = n0 + 16;
    if (out_half) {
      if (n0 < N) { Ch[(size_t)m0 * ldc + n0] = (_Float16)acc00[r];
                    Ch[(size_t)m1 * ldc + n0] = (_Float16)acc10[r]; }
      if (n1 < N) { Ch[(size_t)m0 * ldc + n1] = (_Float16)acc01[r];
                    Ch[(size_t)m1 * ldc + n1] = (_Float16)acc11[r]; }
    } else {
      if (n0 < N) { Cf[(size_t)m0 * ldc + n0] = acc00[r];
                    Cf[(size_t)m1 * ldc + n0] = acc10[r]; }
      if (n1 < N) { Cf[(size_t)m0 * ldc + n1] = acc01[r];
                    Cf[(size_t)m1 * ldc + n1] = acc11[r]; }
    }
  }
}

// ---------------- 4) depthwise causal conv(4) + SiLU ----------------
__global__ void conv_silu(const _Float16* __restrict__ zx, const float* __restrict__ cw,
                          const float* __restrict__ cb, _Float16* __restrict__ xcv) {
  size_t idx = (size_t)blockIdx.x * 256 + threadIdx.x;   // token*640 + ch
  int ch = (int)(idx % CONVD);
  size_t tok = idx / CONVD;
  int t = (int)(tok % TT);
  size_t rowbase = (tok - (size_t)t) * DIP;              // bk*T*DIP
  float acc = cb[ch];
#pragma unroll
  for (int i = 0; i < 4; ++i) {
    int ts = t - 3 + i;
    if (ts >= 0) acc += (float)zx[rowbase + (size_t)ts * DIP + DIN + ch] * cw[ch * 4 + i];
  }
  xcv[idx] = (_Float16)siluf(acc);
}

// ---------------- 5) dt = softplus(raw + bias); dtA = dt * (-exp(A_log)) ----------------
__global__ void dt_prep(const _Float16* __restrict__ zx, const float* __restrict__ dt_bias,
                        const float* __restrict__ A_log, float* __restrict__ dtv,
                        float* __restrict__ dtA) {
  size_t idx = (size_t)blockIdx.x * 256 + threadIdx.x;   // token*8 + h
  int h = (int)(idx & 7);
  size_t tok = idx >> 3;
  float x = (float)zx[tok * DIP + (DIN + CONVD) + h] + dt_bias[h];
  float dt = softplusf(x);
  dtv[idx] = dt;
  dtA[idx] = dt * (-expf(A_log[h]));
}

// ---------------- 6) SSD intra-chunk: Y_diag + chunk states ----------------
__global__ void __launch_bounds__(256)
ssd_chunk(const _Float16* __restrict__ xc, const float* __restrict__ dtv,
          const float* __restrict__ dtA, float* __restrict__ acs_g,
          float* __restrict__ atot_g, float* __restrict__ states,
          float* __restrict__ yacc) {
  __shared__ __align__(16) _Float16 sBC[64 * 128];   // [l][0:64]=B, [l][64:128]=C (contiguous in memory)
  __shared__ _Float16 sX[64 * 64], sXd[64 * 64], sM[64 * 64];
  __shared__ float acs[64];
  const int c = blockIdx.x, h = blockIdx.y, b = blockIdx.z;
  const int tid = threadIdx.x;
  const size_t base = (size_t)b * TT + (size_t)c * 64;

  // async copy of the raw B|C rows: 256B contiguous per position l
  {
    unsigned lds0 = (unsigned)(size_t)(&sBC[0]);
#pragma unroll
    for (int j = 0; j < 4; ++j) {
      int piece = tid * 4 + j;              // 0..1023
      int l = piece >> 4, seg = piece & 15;
      async_b128(xc + (base + l) * CONVD + DIN + seg * 8,
                 lds0 + (unsigned)(l * 128 + seg * 8) * 2u);
    }
  }
  if (tid < 64) acs[tid] = dtA[(base + tid) * 8 + h];
#pragma unroll 4
  for (int i = 0; i < 16; ++i) {
    int idx = tid * 16 + i;                 // l*64 + p
    int l = idx >> 6, p = idx & 63;
    size_t tok = base + l;
    float dt = dtv[tok * 8 + h];
    sX[idx] = (_Float16)((float)xc[tok * CONVD + h * 64 + p] * dt);
  }
  wait_async0();
  __syncthreads();
  if (tid == 0) {                           // short in-LDS inclusive scan
    float r = 0.f;
    for (int l = 0; l < 64; ++l) { r += acs[l]; acs[l] = r; }
    atot_g[((size_t)b * NH + h) * NCH + c] = r;
  }
  __syncthreads();
  float atot = acs[63];
#pragma unroll 4
  for (int i = 0; i < 16; ++i) {
    int idx = tid * 16 + i;
    int l = idx >> 6;
    sXd[idx] = (_Float16)((float)sX[idx] * expf(atot - acs[l]));   // decay-to-end weighted x
  }
  if (tid < 64) acs_g[((size_t)b * NH + h) * TT + c * 64 + tid] = acs[tid];
  __syncthreads();

  const int w = tid >> 5, lane = tid & 31, hi = lane >> 4, lo = lane & 15;
  const int tr = w & 3, tc0 = (w >> 2) * 2;

  // G = C * B^T over state dim (K=64)
  v8f g0 = {}, g1 = {};
  for (int kk = 0; kk < 64; kk += 32) {
    v16h af = frag_rowmajor(sBC + 64, tr * 16 + lo, 128, kk, hi);        // C[l][n]
    v16h b0 = frag_rowmajor(sBC, tc0 * 16 + lo, 128, kk, hi);            // B[s][n]
    v16h b1 = frag_rowmajor(sBC, (tc0 + 1) * 16 + lo, 128, kk, hi);
    g0 = wmma16(af, b0, g0);
    g1 = wmma16(af, b1, g1);
  }
  // apply causal exp(segsum) mask -> M (f16 in LDS)
#pragma unroll
  for (int r = 0; r < 8; ++r) {
    int l = tr * 16 + r + 8 * hi;
    int s0 = tc0 * 16 + lo, s1 = s0 + 16;
    float f0 = (s0 <= l) ? expf(acs[l] - acs[s0]) : 0.f;
    float f1 = (s1 <= l) ? expf(acs[l] - acs[s1]) : 0.f;
    sM[l * 64 + s0] = (_Float16)(g0[r] * f0);
    sM[l * 64 + s1] = (_Float16)(g1[r] * f1);
  }
  __syncthreads();

  // Y_diag = M * X   and   S_chunk = Xd^T * B  (both K = 64 over chunk positions)
  v8f y0 = {}, y1 = {}, st0 = {}, st1 = {};
  for (int kk = 0; kk < 64; kk += 32) {
    v16h am  = frag_rowmajor(sM, tr * 16 + lo, 64, kk, hi);
    v16h bx0 = frag_kmajor(sX, tc0 * 16 + lo, 64, kk, hi);
    v16h bx1 = frag_kmajor(sX, (tc0 + 1) * 16 + lo, 64, kk, hi);
    y0 = wmma16(am, bx0, y0);
    y1 = wmma16(am, bx1, y1);
    v16h ax  = frag_kmajor(sXd, tr * 16 + lo, 64, kk, hi);     // (p, l) via transpose read
    v16h bb0 = frag_kmajor(sBC, tc0 * 16 + lo, 128, kk, hi);   // B[l][n], k-major
    v16h bb1 = frag_kmajor(sBC, (tc0 + 1) * 16 + lo, 128, kk, hi);
    st0 = wmma16(ax, bb0, st0);
    st1 = wmma16(ax, bb1, st1);
  }
  float* Sg = states + (((size_t)b * NCH + c) * NH + h) * (64 * 64);
#pragma unroll
  for (int r = 0; r < 8; ++r) {
    int m  = tr * 16 + r + 8 * hi;          // l for Y, p for S
    int n0 = tc0 * 16 + lo, n1 = n0 + 16;
    yacc[(base + m) * DIN + h * 64 + n0] = y0[r];
    yacc[(base + m) * DIN + h * 64 + n1] = y1[r];
    Sg[m * 64 + n0] = st0[r];
    Sg[m * 64 + n1] = st1[r];
  }
}

// ---------------- 7) inter-chunk scan (in place: states[c] -> state-before-chunk c) ----------------
__global__ void __launch_bounds__(256)
ssd_scan(float* __restrict__ states, const float* __restrict__ atot_g) {
  const int h = blockIdx.x, b = blockIdx.y;
  const int tid = threadIdx.x;
  float S[16];
#pragma unroll
  for (int i = 0; i < 16; ++i) S[i] = 0.f;
  const int off = tid * 16;
  for (int c = 0; c < NCH; ++c) {
    float* p = states + (((size_t)b * NCH + c) * NH + h) * (64 * 64) + off;
    if (c + 1 < NCH)
      __builtin_prefetch((const void*)(p + (size_t)NH * 64 * 64), 1, 1);
    float e = expf(atot_g[((size_t)b * NH + h) * NCH + c]);
#pragma unroll
    for (int i = 0; i < 16; ++i) {
      float cs = p[i];
      p[i] = S[i];                 // P_c : state entering chunk c
      S[i] = e * S[i] + cs;        // S  <- exp(Atot_c)*S + chunkstate_c
    }
  }
}

// ---------------- 8) SSD inter-chunk output: Y_off + D-skip ----------------
__global__ void __launch_bounds__(256)
ssd_off(const _Float16* __restrict__ xc, const float* __restrict__ acs_g,
        const float* __restrict__ states, const float* __restrict__ Dp,
        float* __restrict__ yacc) {
  __shared__ _Float16 sCd[64 * 64], sS[64 * 64];
  const int c = blockIdx.x, h = blockIdx.y, b = blockIdx.z;
  const int tid = threadIdx.x;
  const size_t base = (size_t)b * TT + (size_t)c * 64;
  const float* Sg  = states + (((size_t)b * NCH + c) * NH + h) * (64 * 64);
  const float* acs = acs_g + ((size_t)b * NH + h) * TT + c * 64;

#pragma unroll 4
  for (int i = 0; i < 16; ++i) {
    int idx = tid * 16 + i;
    int l = idx >> 6, n = idx & 63;
    sCd[idx] = (_Float16)((float)xc[(base + l) * CONVD + DIN + DST + n] * expf(acs[l]));
    sS[idx]  = (_Float16)Sg[idx];           // [p][n]
  }
  __syncthreads();

  const int w = tid >> 5, lane = tid & 31, hi = lane >> 4, lo = lane & 15;
  const int tr = w & 3, tc0 = (w >> 2) * 2;
  v8f y0 = {}, y1 = {};
  for (int kk = 0; kk < 64; kk += 32) {     // K = state dim
    v16h af = frag_rowmajor(sCd, tr * 16 + lo, 64, kk, hi);
    v16h b0 = frag_rowmajor(sS, tc0 * 16 + lo, 64, kk, hi);
    v16h b1 = frag_rowmajor(sS, (tc0 + 1) * 16 + lo, 64, kk, hi);
    y0 = wmma16(af, b0, y0);
    y1 = wmma16(af, b1, y1);
  }
  float D = Dp[h];
#pragma unroll
  for (int r = 0; r < 8; ++r) {
    int l  = tr * 16 + r + 8 * hi;
    int p0 = tc0 * 16 + lo, p1 = p0 + 16;
    size_t tok = base + l;
    float xs0 = (float)xc[tok * CONVD + h * 64 + p0];
    float xs1 = (float)xc[tok * CONVD + h * 64 + p1];
    yacc[tok * DIN + h * 64 + p0] += y0[r] + D * xs0;
    yacc[tok * DIN + h * 64 + p1] += y1[r] + D * xs1;
  }
}

// ---------------- 9) gate with silu(z) + RMSNorm -> f16 ----------------
__global__ void __launch_bounds__(256)
gate_norm(const float* __restrict__ yacc, const _Float16* __restrict__ zx,
          const float* __restrict__ norm_w, _Float16* __restrict__ yh) {
  __shared__ float red[256];
  __shared__ float yv[DIN];
  size_t tok = blockIdx.x;
  int tid = threadIdx.x;
  float ssum = 0.f;
#pragma unroll
  for (int j = 0; j < 2; ++j) {
    int d = tid + j * 256;
    float z = (float)zx[tok * DIP + d];
    float y = yacc[tok * DIN + d] * siluf(z);
    yv[d] = y;
    ssum += y * y;
  }
  red[tid] = ssum;
  __syncthreads();
  for (int s = 128; s > 0; s >>= 1) {
    if (tid < s) red[tid] += red[tid + s];
    __syncthreads();
  }
  float rs = rsqrtf(red[0] / (float)DIN + 1e-5f);
#pragma unroll
  for (int j = 0; j < 2; ++j) {
    int d = tid + j * 256;
    yh[tok * DIN + d] = (_Float16)(yv[d] * rs * norm_w[d]);
  }
}

// ---------------- 10) head: q = sum_d(y_out) * softplus(head); logits = q/exp(tau) ----------------
__global__ void __launch_bounds__(256)
head_out(const float* __restrict__ yout, const float* __restrict__ head,
         const float* __restrict__ log_tau, float* __restrict__ out) {
  __shared__ float red[256];
  size_t tok = blockIdx.x;
  int tid = threadIdx.x;
  red[tid] = yout[tok * DM + tid];
  __syncthreads();
  for (int s = 128; s > 0; s >>= 1) {
    if (tid < s) red[tid] += red[tid + s];
    __syncthreads();
  }
  if (tid == 0) {
    int t  = (int)(tok % TT);
    int bk = (int)(tok / TT);
    float q = red[0] * softplusf(head[0]);
    out[(size_t)t * BKB + bk] = q;
    out[(size_t)TOK + (size_t)t * BKB + bk] = q * expf(-log_tau[0]);
  }
}

// ---------------- host side ----------------
extern "C" void kernel_launch(void* const* d_in, const int* in_sizes, int n_in,
                              void* d_out, int out_size, void* d_ws, size_t ws_size,
                              hipStream_t stream) {
  const float* obs        = (const float*)d_in[0];
  const float* W_in       = (const float*)d_in[1];
  const float* b_in       = (const float*)d_in[2];
  const float* in_proj_w  = (const float*)d_in[3];
  const float* conv_w     = (const float*)d_in[4];
  const float* conv_b     = (const float*)d_in[5];
  const float* dt_bias    = (const float*)d_in[6];
  const float* A_log      = (const float*)d_in[7];
  const float* D_param    = (const float*)d_in[8];
  const float* norm_w     = (const float*)d_in[9];
  const float* out_proj_w = (const float*)d_in[10];
  const float* head       = (const float*)d_in[11];
  const float* log_tau    = (const float*)d_in[12];
  float* out = (float*)d_out;

  // workspace carve (256B aligned)
  char* ws = (char*)d_ws;
  size_t off = 0;
  auto carve = [&](size_t bytes) -> char* {
    char* p = ws + off;
    off += (bytes + 255) & ~(size_t)255;
    return p;
  };
  _Float16* u_h   = (_Float16*)carve((size_t)TOK * DM * 2);
  _Float16* wip_h = (_Float16*)carve((size_t)DIP * DM * 2);
  _Float16* wop_h = (_Float16*)carve((size_t)DM * DIN * 2);
  _Float16* zx_h  = (_Float16*)carve((size_t)TOK * DIP * 2);
  _Float16* xcv_h = (_Float16*)carve((size_t)TOK * CONVD * 2);
  float*    dtv   = (float*)carve((size_t)TOK * NH * 4);
  float*    dtA   = (float*)carve((size_t)TOK * NH * 4);
  float*    acs_g = (float*)carve((size_t)BKB * NH * TT * 4);
  float*    atot  = (float*)carve((size_t)BKB * NH * NCH * 4);
  float*    states = (float*)carve((size_t)BKB * NCH * NH * 64 * 64 * 4);
  float*    yacc  = (float*)carve((size_t)TOK * DIN * 4);
  _Float16* y_h   = (_Float16*)carve((size_t)TOK * DIN * 2);
  float*    yout  = (float*)carve((size_t)TOK * DM * 4);

  // 1) weights -> f16
  cvt_weights<<<(DIP * DM + 255) / 256, 256, 0, stream>>>(
      in_proj_w, wip_h, DIP * DM, out_proj_w, wop_h, DM * DIN);
  // 2) embed
  embed_k<<<TOK, 256, 0, stream>>>(obs, W_in, b_in, u_h);
  // 3) in_proj GEMM: (TOK x 1160) = u (TOK x 256) * W^T
  gemm_wmma<<<dim3((DIP + 63) / 64, TOK / 128), 256, 0, stream>>>(
      u_h, wip_h, nullptr, zx_h, TOK, DIP, DM, DIP, 1);
  // 4) conv + silu
  conv_silu<<<(size_t)TOK * CONVD / 256, 256, 0, stream>>>(zx_h, conv_w, conv_b, xcv_h);
  // 5) dt
  dt_prep<<<(size_t)TOK * NH / 256, 256, 0, stream>>>(zx_h, dt_bias, A_log, dtv, dtA);
  // 6) SSD intra-chunk
  ssd_chunk<<<dim3(NCH, NH, BKB), 256, 0, stream>>>(xcv_h, dtv, dtA, acs_g, atot, states, yacc);
  // 7) inter-chunk state scan
  ssd_scan<<<dim3(NH, BKB), 256, 0, stream>>>(states, atot);
  // 8) inter-chunk output + D skip
  ssd_off<<<dim3(NCH, NH, BKB), 256, 0, stream>>>(xcv_h, acs_g, states, D_param, yacc);
  // 9) gate + RMSNorm
  gate_norm<<<TOK, 256, 0, stream>>>(yacc, zx_h, norm_w, y_h);
  // 10) out_proj GEMM: (TOK x 256) = y (TOK x 512) * W^T
  gemm_wmma<<<dim3(DM / 64, TOK / 128), 256, 0, stream>>>(
      y_h, wop_h, yout, nullptr, TOK, DM, DIN, DM, 0);
  // 11) head reduction -> q, logits
  head_out<<<TOK, 256, 0, stream>>>(yout, head, log_tau, out);
}
